// GPNStarRowSelfAttention_5789615915686
// MI455X (gfx1250) — matrix operand order repaired and checked by reading, not compiled, not run on hardware
//
#include <hip/hip_runtime.h>
#include <hip/hip_bf16.h>
#include <math.h>

// Problem constants (from reference)
#define B_   2
#define R_   64
#define L_   512
#define H_   1024
#define NH_  8
#define HD_  64
#define AH_  512   // NH_*HD_
#define HD2_ 32

typedef _Float16 v16h __attribute__((ext_vector_type(16)));
typedef _Float16 v8h  __attribute__((ext_vector_type(8)));
typedef float    v8f  __attribute__((ext_vector_type(8)));

// ---------------------------------------------------------------------------
// WMMA helper: D = A(16x32 f16) * B(32x16 f16) + C (f32 accum)
// ---------------------------------------------------------------------------
__device__ __forceinline__ v8f wmma_f16(v16h a, v16h b, v8f c) {
  return __builtin_amdgcn_wmma_f32_16x16x32_f16(
      /*neg_a=*/false, a, /*neg_b=*/false, b,
      /*c_mod=*/(short)0, c, /*reuse_a=*/false, /*reuse_b=*/false);
}

// ---------------------------------------------------------------------------
// Fragment loaders (CDNA5 wave32 WMMA VGPR layouts, CDNA5 ISA)
// A (16x32, f16): lane<16 -> row=lane, K in {0..7, 16..23}; lane>=16 ->
//                 row=lane-16, K in {8..15, 24..31}.  Two 16B chunks per lane.
// B (32x16, f16): lane<16 -> col=lane, K in {0..7, 16..23}; lane>=16 -> col=lane-16,
//                 K in {8..15, 24..31}.  Two 16B chunks per lane; we read B as
//                 rows of srcT, i.e. B[k][n] = src[n][k] (src row-major N x K).
// ---------------------------------------------------------------------------
__device__ __forceinline__ v16h load_a_f32(const float* __restrict__ src, int ld,
                                           int row0, int k0, int lane) {
  int r = lane & 15, kh = lane >> 4;
  const float* p = src + (size_t)(row0 + r) * ld + k0 + kh * 8;
  v16h a;
#pragma unroll
  for (int i = 0; i < 8; ++i) {
    a[i]     = (_Float16)p[i];        // K = kh*8 + i
    a[i + 8] = (_Float16)p[i + 16];   // K = 16 + kh*8 + i
  }
  return a;
}

__device__ __forceinline__ v16h load_a_f16(const _Float16* __restrict__ src, int ld,
                                           int row0, int k0, int lane) {
  int r = lane & 15, kh = lane >> 4;
  const _Float16* p = src + (size_t)(row0 + r) * ld + k0 + kh * 8;
  v8h lo = *(const v8h*)p;
  v8h hi = *(const v8h*)(p + 16);
  v16h a;
#pragma unroll
  for (int i = 0; i < 8; ++i) { a[i] = lo[i]; a[i + 8] = hi[i]; }
  return a;
}

__device__ __forceinline__ v16h load_bT_f16(const _Float16* __restrict__ src, int ld,
                                            int n0, int k0, int lane) {
  int n = lane & 15, kh = lane >> 4;
  const _Float16* p = src + (size_t)(n0 + n) * ld + k0 + kh * 8;
  v8h lo = *(const v8h*)p;
  v8h hi = *(const v8h*)(p + 16);
  v16h b;
#pragma unroll
  for (int i = 0; i < 8; ++i) { b[i] = lo[i]; b[i + 8] = hi[i]; }
  return b;
}

// ---------------------------------------------------------------------------
// Kernel 0: one-shot weight f32 -> f16 pre-convert (removes v_cvt from the
// B-fragment path of every GEMM hot loop; weights are L2-resident afterwards).
// ---------------------------------------------------------------------------
__global__ __launch_bounds__(256) void cvt_w_kernel(
    const float* __restrict__ Wq, const float* __restrict__ Wk,
    const float* __restrict__ Wv, _Float16* __restrict__ wq16,
    _Float16* __restrict__ wk16, _Float16* __restrict__ wv16) {
  int t = blockIdx.x * blockDim.x + threadIdx.x;
  const int N = AH_ * H_;
  if (t >= 3 * N) return;
  const float* s = (t < N) ? Wq : ((t < 2 * N) ? Wk : Wv);
  _Float16*    d = (t < N) ? wq16 : ((t < 2 * N) ? wk16 : wv16);
  int u = (t < N) ? t : ((t < 2 * N) ? t - N : t - 2 * N);
  d[u] = (_Float16)s[u];
}

// ---------------------------------------------------------------------------
// Kernel 1: Q/K projection.  C[m][n] = row0[m][:] . W[n][:] + bias[n]
// M = B*L = 1024, N = AH = 512, K = H = 1024.  grid (N/256, M/128, 2), block 256.
// Wave tile 64x64 (4x4 fragments), waves arranged 2(M) x 4(N).
// ---------------------------------------------------------------------------
__global__ __launch_bounds__(256) void gemm_qk_kernel(
    const float* __restrict__ hs, const _Float16* __restrict__ wq16,
    const float* __restrict__ bq, const _Float16* __restrict__ wk16,
    const float* __restrict__ bk, float* __restrict__ qraw,
    float* __restrict__ kraw) {
  const _Float16* W  = (blockIdx.z == 0) ? wq16 : wk16;
  const float* bias  = (blockIdx.z == 0) ? bq : bk;
  float* out         = (blockIdx.z == 0) ? qraw : kraw;

  const int lane = threadIdx.x & 31;
  const int wave = threadIdx.x >> 5;
  const int wm = wave >> 2, wn = wave & 3;
  const int mb = blockIdx.y * 128;            // global row base (m = b*L + l)
  const int n0 = blockIdx.x * 256 + wn * 64;  // global col base for this wave
  const int bb = mb / L_;                     // batch index (128 | 512, no straddle)
  const int l0 = mb % L_;
  // row0 rows are hs[b][0][l][:]  ->  contiguous with stride H starting at l0
  const float* Abase = hs + ((size_t)bb * R_ * L_ + l0) * H_;
  const int mloc = wm * 64;

  v8f acc[4][4];
#pragma unroll
  for (int tj = 0; tj < 4; ++tj) {
    float bv = bias[n0 + tj * 16 + (lane & 15)];  // fixed column per lane
#pragma unroll
    for (int ti = 0; ti < 4; ++ti)
#pragma unroll
      for (int e = 0; e < 8; ++e) acc[ti][tj][e] = bv;
  }

  for (int k0 = 0; k0 < H_; k0 += 32) {
    v16h a[4], bm[4];
#pragma unroll
    for (int ti = 0; ti < 4; ++ti) a[ti]  = load_a_f32(Abase, H_, mloc + ti * 16, k0, lane);
#pragma unroll
    for (int tj = 0; tj < 4; ++tj) bm[tj] = load_bT_f16(W, H_, n0 + tj * 16, k0, lane);
#pragma unroll
    for (int ti = 0; ti < 4; ++ti)
#pragma unroll
      for (int tj = 0; tj < 4; ++tj) acc[ti][tj] = wmma_f16(a[ti], bm[tj], acc[ti][tj]);
  }

  // Store C (f32, row-major [B*L][AH]): VGPR e -> M = e + 8*(lane>>4), N = lane&15
  const int n = lane & 15, kh = lane >> 4;
#pragma unroll
  for (int ti = 0; ti < 4; ++ti) {
    int mrow = mb + mloc + ti * 16 + 8 * kh;
#pragma unroll
    for (int tj = 0; tj < 4; ++tj) {
      int col = n0 + tj * 16 + n;
#pragma unroll
      for (int e = 0; e < 8; ++e)
        out[(size_t)(mrow + e) * AH_ + col] = acc[ti][tj][e];
    }
  }
}

// ---------------------------------------------------------------------------
// Kernel 2: RoPE + f32 -> f16 repack into head-major q16/k16 [B][NH][L][HD]
// ---------------------------------------------------------------------------
__global__ __launch_bounds__(256) void rope_kernel(
    const float* __restrict__ qraw, const float* __restrict__ kraw,
    const float* __restrict__ sp, _Float16* __restrict__ q16,
    _Float16* __restrict__ k16) {
  int t = blockIdx.x * blockDim.x + threadIdx.x;
  const int total = B_ * NH_ * L_ * HD2_;
  if (t >= 2 * total) return;
  const float* src = (t < total) ? qraw : kraw;
  _Float16*    dst = (t < total) ? q16 : k16;
  int u = (t < total) ? t : t - total;
  int i = u % HD2_; u /= HD2_;
  int l = u % L_;   u /= L_;
  int h = u % NH_;
  int b = u / NH_;
  float s = sp[l * HD_ + i];         // sin half
  float c = sp[l * HD_ + HD2_ + i];  // cos half
  const float* pr = src + ((size_t)(b * L_ + l)) * AH_ + h * HD_ + 2 * i;
  float x0 = pr[0], x1 = pr[1];
  _Float16* pw = dst + (((size_t)(b * NH_ + h) * L_ + l)) * HD_ + 2 * i;
  pw[0] = (_Float16)(x0 * c - x1 * s);
  pw[1] = (_Float16)(x1 * c + x0 * s);
}

// ---------------------------------------------------------------------------
// Kernel 3: V projection, stored TRANSPOSED as f16:
//   vT[b][r][h][d][l] = hs[b][r][l][:] . Wv[h*HD+d][:] + bv
// M = B*R*L = 65536, N = 512, K = 1024.  grid (2, 512), block 256.
// C-frag per lane has fixed N (=d) and 8 consecutive M (=l) -> one b128 store.
// ---------------------------------------------------------------------------
__global__ __launch_bounds__(256) void gemm_v_kernel(
    const float* __restrict__ hs, const _Float16* __restrict__ wv16,
    const float* __restrict__ bv, _Float16* __restrict__ vT) {
  const int lane = threadIdx.x & 31;
  const int wave = threadIdx.x >> 5;
  const int wm = wave >> 2, wn = wave & 3;
  const int mb = blockIdx.y * 128;
  const int n0 = blockIdx.x * 256 + wn * 64;
  const float* Abase = hs + (size_t)mb * H_;  // flat rows of [B][R][L][H]
  const int mloc = wm * 64;
  // decompose block row base (128 | 512: tile stays inside one (b,r))
  const int b = mb / (R_ * L_);
  const int r = (mb / L_) % R_;
  const int l0 = mb % L_;

  v8f acc[4][4];
#pragma unroll
  for (int tj = 0; tj < 4; ++tj) {
    float bvv = bv[n0 + tj * 16 + (lane & 15)];
#pragma unroll
    for (int ti = 0; ti < 4; ++ti)
#pragma unroll
      for (int e = 0; e < 8; ++e) acc[ti][tj][e] = bvv;
  }

  for (int k0 = 0; k0 < H_; k0 += 32) {
    // prefetch next k-slab of activations (global_prefetch_b8)
    if (k0 + 32 < H_)
      __builtin_prefetch(Abase + (size_t)(mloc + (lane & 15)) * H_ + k0 + 32, 0, 1);
    v16h a[4], bm[4];
#pragma unroll
    for (int ti = 0; ti < 4; ++ti) a[ti]  = load_a_f32(Abase, H_, mloc + ti * 16, k0, lane);
#pragma unroll
    for (int tj = 0; tj < 4; ++tj) bm[tj] = load_bT_f16(wv16, H_, n0 + tj * 16, k0, lane);
#pragma unroll
    for (int ti = 0; ti < 4; ++ti)
#pragma unroll
      for (int tj = 0; tj < 4; ++tj) acc[ti][tj] = wmma_f16(a[ti], bm[tj], acc[ti][tj]);
  }

  const int n = lane & 15, kh = lane >> 4;
#pragma unroll
  for (int ti = 0; ti < 4; ++ti) {
    int lbase = l0 + mloc + ti * 16 + 8 * kh;  // 8 consecutive l values
#pragma unroll
    for (int tj = 0; tj < 4; ++tj) {
      int col = n0 + tj * 16 + n;  // global output channel
      int hh = col >> 6;           // / HD
      int dd = col & 63;           // % HD
      v8h pk;
#pragma unroll
      for (int e = 0; e < 8; ++e) pk[e] = (_Float16)acc[ti][tj][e];
      _Float16* dst =
          vT + ((((size_t)(b * R_ + r) * NH_ + hh) * HD_ + dd) * L_) + lbase;
      *(v8h*)dst = pk;  // 16B packed store
    }
  }
}

// ---------------------------------------------------------------------------
// Kernel 4: scores = q k^T / 8, softmax over keys, probs -> f16.
// grid (B*NH, L/16); block 256 = 8 waves; each wave does 16(q) x 64(keys),
// K = HD = 64 (2 WMMA k-steps). Scores staged in LDS, two-stage row reduce.
// ---------------------------------------------------------------------------
__global__ __launch_bounds__(256) void scores_softmax_kernel(
    const _Float16* __restrict__ q16, const _Float16* __restrict__ k16,
    _Float16* __restrict__ probs) {
  const int bh = blockIdx.x;
  const int q0 = blockIdx.y * 16;
  const _Float16* Q = q16 + (size_t)bh * L_ * HD_;
  const _Float16* Kp = k16 + (size_t)bh * L_ * HD_;

  __shared__ float S[16][L_ + 8];
  __shared__ float red[16][16];

  const int lane = threadIdx.x & 31;
  const int wave = threadIdx.x >> 5;
  const int n0 = wave * 64;

  v8f acc[4] = {};
#pragma unroll
  for (int k0 = 0; k0 < HD_; k0 += 32) {
    v16h a = load_a_f16(Q, HD_, q0, k0, lane);
#pragma unroll
    for (int tj = 0; tj < 4; ++tj) {
      v16h bm = load_bT_f16(Kp, HD_, n0 + tj * 16, k0, lane);
      acc[tj] = wmma_f16(a, bm, acc[tj]);
    }
  }

  const float scale = 0.125f;  // 1/sqrt(HD)
  const int n = lane & 15, kh = lane >> 4;
#pragma unroll
  for (int tj = 0; tj < 4; ++tj)
#pragma unroll
    for (int e = 0; e < 8; ++e)
      S[e + 8 * kh][n0 + tj * 16 + n] = acc[tj][e] * scale;
  __syncthreads();

  // softmax: 16 threads per row, each owns 32 strided columns
  const int row = threadIdx.x >> 4;
  const int c0 = threadIdx.x & 15;
  float pmax = -1e30f;
#pragma unroll 8
  for (int j = 0; j < 32; ++j) pmax = fmaxf(pmax, S[row][c0 + 16 * j]);
  red[row][c0] = pmax;
  __syncthreads();
  float rmax = -1e30f;
#pragma unroll
  for (int j = 0; j < 16; ++j) rmax = fmaxf(rmax, red[row][j]);
  float psum = 0.f;
#pragma unroll 8
  for (int j = 0; j < 32; ++j) {
    float e = __expf(S[row][c0 + 16 * j] - rmax);
    S[row][c0 + 16 * j] = e;
    psum += e;
  }
  __syncthreads();  // everyone done reading red(max) before overwrite
  red[row][c0] = psum;
  __syncthreads();
  float rsum = 0.f;
#pragma unroll
  for (int j = 0; j < 16; ++j) rsum += red[row][j];
  float inv = 1.f / rsum;

  _Float16* P = probs + (size_t)bh * L_ * L_ + (size_t)(q0 + row) * L_;
#pragma unroll 8
  for (int j = 0; j < 32; ++j)
    P[c0 + 16 * j] = (_Float16)(S[row][c0 + 16 * j] * inv);
}

// ---------------------------------------------------------------------------
// Kernel 5: ctx[b,r,h] = probs[b,h] (512x512) @ v[b,r,h] (512x64), f32 out.
// grid (B*R*NH); block 256 = 8 waves, wave w covers q-rows [w*64, w*64+64).
// The 64KB V tile (vT[b][r][h], contiguous HD x L f16) is shared by all 8
// waves, so stage it once per block into LDS with gfx1250 async load-to-LDS
// (ASYNCcnt path), then B-fragments are contiguous 16B ds_load_b128 reads.
// Saves ~8x global traffic on the 34-GFLOP einsum.
// ---------------------------------------------------------------------------
__global__ __launch_bounds__(256) void ctx_kernel(
    const _Float16* __restrict__ probs, const _Float16* __restrict__ vT,
    float* __restrict__ out) {
  const int idx = blockIdx.x;          // (b*R + r)*NH + h
  const int h = idx % NH_;
  const int br = idx / NH_;            // b*R + r
  const int b = br / R_;
  const int lane = threadIdx.x & 31;
  const int wave = threadIdx.x >> 5;
  const int m0 = wave * 64;

  const _Float16* P = probs + (size_t)(b * NH_ + h) * L_ * L_;
  const _Float16* Vg = vT + (size_t)idx * HD_ * L_;
  float* O = out + (size_t)br * L_ * AH_ + h * HD_;

  __shared__ _Float16 Vs[HD_ * L_];  // 64KB, linear copy of vT[b][r][h]

  // ---- async stage: 256 threads x 16 iters x 16B = 64KB ----
  {
    const uint32_t ldsbase = (uint32_t)(uintptr_t)(&Vs[0]);
#pragma unroll
    for (int it = 0; it < 16; ++it) {
      uint32_t off = ((uint32_t)threadIdx.x + (uint32_t)it * 256u) * 16u;  // bytes
      uint64_t ga  = (uint64_t)(uintptr_t)((const char*)Vg + off);
      uint32_t la  = ldsbase + off;
      asm volatile("global_load_async_to_lds_b128 %0, %1, off"
                   :
                   : "v"(la), "v"(ga)
                   : "memory");
    }
    asm volatile("s_wait_asynccnt 0" ::: "memory");
  }
  __syncthreads();

  v8f acc[4][4] = {};
  for (int k0 = 0; k0 < L_; k0 += 32) {
    v16h a[4], bm[4];
#pragma unroll
    for (int ti = 0; ti < 4; ++ti) a[ti]  = load_a_f16(P, L_, m0 + ti * 16, k0, lane);
#pragma unroll
    for (int tj = 0; tj < 4; ++tj) bm[tj] = load_bT_f16(Vs, L_, tj * 16, k0, lane);
#pragma unroll
    for (int ti = 0; ti < 4; ++ti)
#pragma unroll
      for (int tj = 0; tj < 4; ++tj) acc[ti][tj] = wmma_f16(a[ti], bm[tj], acc[ti][tj]);
  }

  const int n = lane & 15, kh = lane >> 4;
#pragma unroll
  for (int ti = 0; ti < 4; ++ti) {
    int mrow = m0 + ti * 16 + 8 * kh;
#pragma unroll
    for (int tj = 0; tj < 4; ++tj) {
      int col = tj * 16 + n;
#pragma unroll
      for (int e = 0; e < 8; ++e)
        O[(size_t)(mrow + e) * AH_ + col] = acc[ti][tj][e];
    }
  }
}

// ---------------------------------------------------------------------------
// Launcher.  Workspace layout (bytes):
//   [0, 2M)    qraw  f32 [B][L][AH]
//   [2M, 4M)   kraw  f32 [B][L][AH]
//   [4M, 5M)   q16   f16 [B][NH][L][HD]
//   [5M, 6M)   k16   f16 [B][NH][L][HD]
//   [6M, 14M)  probs f16 [B][NH][L][L]
//   [14M, 78M) vT    f16 [B][R][NH][HD][L]
//   [78M, 79M) wq16  f16 [AH][H]
//   [79M, 80M) wk16  f16 [AH][H]
//   [80M, 81M) wv16  f16 [AH][H]
// ---------------------------------------------------------------------------
extern "C" void kernel_launch(void* const* d_in, const int* in_sizes, int n_in,
                              void* d_out, int out_size, void* d_ws,
                              size_t ws_size, hipStream_t stream) {
  const float* hs = (const float*)d_in[0];
  const float* sp = (const float*)d_in[1];
  const float* Wq = (const float*)d_in[2];
  const float* bq = (const float*)d_in[3];
  const float* Wk = (const float*)d_in[4];
  const float* bk = (const float*)d_in[5];
  const float* Wv = (const float*)d_in[6];
  const float* bv = (const float*)d_in[7];
  float* out = (float*)d_out;

  char* ws = (char*)d_ws;
  const size_t MB = 1024u * 1024u;
  float*    qraw  = (float*)(ws);
  float*    kraw  = (float*)(ws + 2 * MB);
  _Float16* q16   = (_Float16*)(ws + 4 * MB);
  _Float16* k16   = (_Float16*)(ws + 5 * MB);
  _Float16* probs = (_Float16*)(ws + 6 * MB);
  _Float16* vT    = (_Float16*)(ws + 14 * MB);
  _Float16* wq16  = (_Float16*)(ws + 78 * MB);
  _Float16* wk16  = (_Float16*)(ws + 79 * MB);
  _Float16* wv16  = (_Float16*)(ws + 80 * MB);

  // 0) weight pre-convert f32 -> f16 (1.5M elems total)
  {
    int total = 3 * AH_ * H_;
    cvt_w_kernel<<<(total + 255) / 256, 256, 0, stream>>>(Wq, Wk, Wv, wq16, wk16, wv16);
  }
  // 1) Q/K projection (z=0 -> Q, z=1 -> K)
  gemm_qk_kernel<<<dim3(AH_ / 256, (B_ * L_) / 128, 2), 256, 0, stream>>>(
      hs, wq16, bq, wk16, bk, qraw, kraw);
  // 2) RoPE + pack f16
  {
    int total = 2 * B_ * NH_ * L_ * HD2_;
    rope_kernel<<<(total + 255) / 256, 256, 0, stream>>>(qraw, kraw, sp, q16, k16);
  }
  // 3) V projection -> transposed f16
  gemm_v_kernel<<<dim3(AH_ / 256, (B_ * R_ * L_) / 128), 256, 0, stream>>>(
      hs, wv16, bv, vT);
  // 4) scores + softmax -> probs
  scores_softmax_kernel<<<dim3(B_ * NH_, L_ / 16), 256, 0, stream>>>(q16, k16, probs);
  // 5) ctx = probs @ v -> output (async LDS staging of V)
  ctx_kernel<<<B_ * R_ * NH_, 256, 0, stream>>>(probs, vT, out);
}
